// MultiHeadAttentionSocial_1821066134213
// MI455X (gfx1250) — matrix operand order repaired and checked by reading, not compiled, not run on hardware
//
#include <hip/hip_runtime.h>
#include <hip/hip_bf16.h>

// ---------------------------------------------------------------------------
// MultiHeadAttentionSocial — CDNA5 (gfx1250) WMMA implementation.
// B=4, S=Skv=2048, H=16, Dk=64, D_MODEL=1024, USER_DIM=512.
// GEMMs + attention via v_wmma_f32_16x16x32_bf16; attention K tiles staged
// through the Tensor Data Mover (tensor_load_to_lds, TENSORcnt).
// ---------------------------------------------------------------------------

#define B_   4
#define S_   2048
#define DM_  1024
#define UD_  512
#define H_   16
#define DK_  64
#define MTOT (B_ * S_)          // 8192 rows

typedef __attribute__((ext_vector_type(16))) __bf16 v16bf;
typedef __attribute__((ext_vector_type(8)))  float  v8f;
typedef unsigned int v4u __attribute__((ext_vector_type(4)));
typedef unsigned int v8u __attribute__((ext_vector_type(8)));

union FragAB { uint4 u4[2]; v16bf v; };
union U4     { uint4 u4; unsigned short us[8]; };

__device__ inline unsigned short f2bf(float x) {
    unsigned u = __float_as_uint(x);
    unsigned r = (u + 0x7FFFu + ((u >> 16) & 1u)) >> 16;
    return (unsigned short)r;
}

__device__ inline v8f v8zero() {
    v8f z;
#pragma unroll
    for (int i = 0; i < 8; ++i) z[i] = 0.0f;
    return z;
}

// ---------------------------------------------------------------------------
// TDM: async 2D tile load global -> LDS via Tensor Data Mover.
// Tile: 32 rows x 64 cols of bf16 (2B), row stride 1024 elements.
// D# built per CDNA5 ISA; groups 0/1 only (2D tensor).
// ---------------------------------------------------------------------------
__device__ inline void tdm_load_k_tile(const unsigned short* gptr, unsigned lds_off) {
    unsigned long long ga = (unsigned long long)(uintptr_t)gptr;
    v4u g0;
    g0[0] = 1u;                                           // count=1, user mode
    g0[1] = lds_off;                                      // lds_addr (bytes)
    g0[2] = (unsigned)(ga & 0xffffffffu);                 // global_addr[31:0]
    g0[3] = (unsigned)((ga >> 32) & 0x01ffffffu)          // global_addr[56:32]
            | (2u << 30);                                 // type=2 ("image")
    v8u g1;
    g1[0] = (1u << 16);                                   // data_size=1 -> 2B
    g1[1] = (64u << 16);                                  // tensor_dim0 = 64
    g1[2] = (32u << 16);                                  // tensor_dim1 = 32
    g1[3] = (64u << 16);                                  // tile_dim0 = 64
    g1[4] = 32u;                                          // tile_dim1=32, tile_dim2=0
    g1[5] = 1024u;                                        // tensor_dim0_stride lo
    g1[6] = 0u;                                           // stride hi / dim1_stride lo
    g1[7] = 0u;
    asm volatile("tensor_load_to_lds %0, %1" :: "s"(g0), "s"(g1) : "memory");
}

// ---------------------------------------------------------------------------
// fp32 -> bf16 bulk convert (weights)
// ---------------------------------------------------------------------------
__global__ void __launch_bounds__(256) cvt_f32_bf16(const float* __restrict__ in,
                                                    unsigned short* __restrict__ out,
                                                    int n4) {
    int i = blockIdx.x * 256 + threadIdx.x;
    if (i < n4) {
        float4 f = ((const float4*)in)[i];
        unsigned short* d = out + (size_t)i * 4;
        d[0] = f2bf(f.x); d[1] = f2bf(f.y); d[2] = f2bf(f.z); d[3] = f2bf(f.w);
    }
}

// ---------------------------------------------------------------------------
// zero-row masks: mask[row] = (sum(raw row) == 0), one block per row
// ---------------------------------------------------------------------------
__global__ void __launch_bounds__(256) mask_kernel(const float* __restrict__ key,
                                                   const float* __restrict__ value,
                                                   int* __restrict__ kmask,
                                                   int* __restrict__ vmask) {
    __shared__ float2 red[256];
    const int tid = threadIdx.x;
    const size_t row = blockIdx.x;
    const float* kr = key   + row * UD_;
    const float* vr = value + row * UD_;
    float2 acc;
    acc.x = kr[tid] + kr[tid + 256];
    acc.y = vr[tid] + vr[tid + 256];
    red[tid] = acc;
    __syncthreads();
#pragma unroll
    for (int s = 128; s > 0; s >>= 1) {
        if (tid < s) { red[tid].x += red[tid + s].x; red[tid].y += red[tid + s].y; }
        __syncthreads();
    }
    if (tid == 0) {
        kmask[row] = (red[0].x == 0.0f) ? 1 : 0;
        vmask[row] = (red[0].y == 0.0f) ? 1 : 0;
    }
}

// ---------------------------------------------------------------------------
// GEMM: C[M,1024] = A[M,K] * W[K,1024] + bias, optional row-mask zeroing.
// Block 256 thr (8 waves), tile 128x128, BK=32; wave = 32x64 (2x4 WMMA tiles)
// ---------------------------------------------------------------------------
template <bool A_BF16, bool OUT_F32>
__global__ void __launch_bounds__(256) gemm_proj(const void* __restrict__ Aptr, int K,
                                                 const unsigned short* __restrict__ Wb,
                                                 const float* __restrict__ bias,
                                                 const int* __restrict__ mask,
                                                 void* __restrict__ Cptr) {
    __shared__ __align__(16) unsigned short Alds[128 * 32];      // [m][k]
    __shared__ __align__(16) unsigned short Blds[128 * 32];      // [n][k] (transposed)

    const int tid  = threadIdx.x;
    const int lane = tid & 31;
    const int wave = tid >> 5;
    const int waveM = wave >> 1;      // 0..3
    const int waveN = wave & 1;       // 0..1
    const int lr = lane & 15;
    const int lg = lane >> 4;
    const int m0 = blockIdx.x * 128;
    const int n0 = blockIdx.y * 128;

    v8f acc[2][4];
#pragma unroll
    for (int i = 0; i < 2; ++i)
#pragma unroll
        for (int j = 0; j < 4; ++j) acc[i][j] = v8zero();

    for (int k0 = 0; k0 < K; k0 += 32) {
        // ---- stage A tile 128x32 into LDS (bf16) ----
        if (A_BF16) {
            const unsigned short* A = (const unsigned short*)Aptr;
#pragma unroll
            for (int i = 0; i < 2; ++i) {
                int idx = tid + i * 256;          // 512 vecs of 8 bf16
                int row = idx >> 2;
                int col = (idx & 3) * 8;
                *(uint4*)&Alds[row * 32 + col] =
                    *(const uint4*)(A + (size_t)(m0 + row) * K + k0 + col);
            }
        } else {
            const float* A = (const float*)Aptr;
#pragma unroll
            for (int i = 0; i < 4; ++i) {
                int idx = tid + i * 256;          // 1024 vecs of 4 fp32
                int row = idx >> 3;
                int col = (idx & 7) * 4;
                float4 f = *(const float4*)(A + (size_t)(m0 + row) * K + k0 + col);
                unsigned short* d = &Alds[row * 32 + col];
                d[0] = f2bf(f.x); d[1] = f2bf(f.y); d[2] = f2bf(f.z); d[3] = f2bf(f.w);
            }
        }
        // ---- stage W tile 32x128, transposed into LDS as [n][k] ----
#pragma unroll
        for (int i = 0; i < 2; ++i) {
            int idx  = tid + i * 256;             // 512 vecs of 8 bf16
            int krow = idx >> 4;
            int ncol = (idx & 15) * 8;
            U4 w;
            w.u4 = *(const uint4*)(Wb + (size_t)(k0 + krow) * 1024 + n0 + ncol);
#pragma unroll
            for (int e = 0; e < 8; ++e) Blds[(ncol + e) * 32 + krow] = w.us[e];
        }
        __syncthreads();

        // ---- fragments + WMMA ----
        FragAB af[2], bfr[4];
#pragma unroll
        for (int mt = 0; mt < 2; ++mt) {
            const unsigned short* base = &Alds[(waveM * 32 + mt * 16 + lr) * 32 + lg * 8];
            af[mt].u4[0] = *(const uint4*)(base);
            af[mt].u4[1] = *(const uint4*)(base + 16);
        }
#pragma unroll
        for (int nt = 0; nt < 4; ++nt) {
            const unsigned short* base = &Blds[(waveN * 64 + nt * 16 + lr) * 32 + lg * 8];
            bfr[nt].u4[0] = *(const uint4*)(base);
            bfr[nt].u4[1] = *(const uint4*)(base + 16);
        }
#pragma unroll
        for (int mt = 0; mt < 2; ++mt)
#pragma unroll
            for (int nt = 0; nt < 4; ++nt)
                acc[mt][nt] = __builtin_amdgcn_wmma_f32_16x16x32_bf16(
                    false, af[mt].v, false, bfr[nt].v, (short)0, acc[mt][nt], false, false);
        __syncthreads();
    }

    // ---- epilogue: bias, mask, store ----
#pragma unroll
    for (int mt = 0; mt < 2; ++mt) {
#pragma unroll
        for (int nt = 0; nt < 4; ++nt) {
            int col = n0 + waveN * 64 + nt * 16 + lr;
            float bv = bias[col];
#pragma unroll
            for (int j = 0; j < 8; ++j) {
                int row = m0 + waveM * 32 + mt * 16 + j + lg * 8;
                float v = acc[mt][nt][j] + bv;
                if (mask && mask[row]) v = 0.0f;
                if (OUT_F32)
                    ((float*)Cptr)[(size_t)row * 1024 + col] = v;
                else
                    ((unsigned short*)Cptr)[(size_t)row * 1024 + col] = f2bf(v);
            }
        }
    }
}

// ---------------------------------------------------------------------------
// Flash attention. Grid: (S/128, B*H). Block 256 thr; wave w owns q rows
// [blk*128 + w*16, +16). KV processed 32 rows/step, double-buffered:
//   K tile  : TDM tensor_load_to_lds (wave 0 issues, TENSORcnt drain)
//   V tile  : cooperative load, transposed into LDS through VGPRs
// Online softmax in f32; P goes C-layout -> A-layout via per-wave LDS.
// ---------------------------------------------------------------------------
__global__ void __launch_bounds__(256) attn_kernel(const unsigned short* __restrict__ qb,
                                                   const unsigned short* __restrict__ kb,
                                                   const unsigned short* __restrict__ vb,
                                                   unsigned short* __restrict__ ob) {
    __shared__ __align__(16) unsigned short klds[2][32 * 64];    // [kv][d]
    __shared__ __align__(16) unsigned short vT[2][64 * 32];      // [d][kv]
    __shared__ __align__(16) unsigned short Plds[8][16 * 32];    // per-wave P

    const int tid  = threadIdx.x;
    const int lane = tid & 31;
    const int wave = tid >> 5;
    const int lr = lane & 15;
    const int lg = lane >> 4;
    const int h = blockIdx.y & 15;
    const int b = blockIdx.y >> 4;
    const int q0 = blockIdx.x * 128 + wave * 16;
    const size_t qrow = (size_t)b * S_ + q0;
    const int hcol = h * DK_;
    const size_t kvbase = (size_t)b * S_;

    // resident q A-fragments (Dk=64 -> two K=32 slices)
    FragAB qf[2];
#pragma unroll
    for (int dsl = 0; dsl < 2; ++dsl) {
        const unsigned short* base = qb + (qrow + lr) * DM_ + hcol + dsl * 32 + lg * 8;
        qf[dsl].u4[0] = *(const uint4*)(base);
        qf[dsl].u4[1] = *(const uint4*)(base + 16);
    }

    float mrow[8], lrow[8];
    v8f o[4];
#pragma unroll
    for (int j = 0; j < 8; ++j) { mrow[j] = -1e30f; lrow[j] = 0.0f; }
#pragma unroll
    for (int dt = 0; dt < 4; ++dt) o[dt] = v8zero();

    const int ldr = tid >> 3;          // cooperative-load row 0..31
    const int ldc = (tid & 7) * 8;     // cooperative-load col

    // ---- prologue: stage tile 0 into buffer 0 ----
    if (wave == 0)
        tdm_load_k_tile(kb + (kvbase + 0) * DM_ + hcol,
                        (unsigned)(uintptr_t)&klds[0][0]);
    {
        U4 vv;
        vv.u4 = *(const uint4*)(vb + (kvbase + 0 + ldr) * DM_ + hcol + ldc);
#pragma unroll
        for (int e = 0; e < 8; ++e) vT[0][(ldc + e) * 32 + ldr] = vv.us[e];
    }
    if (wave == 0) __builtin_amdgcn_s_wait_tensorcnt(0);
    __syncthreads();

    const int NT = S_ / 32;
    for (int it = 0; it < NT; ++it) {
        const int buf = it & 1;
        const int nbuf = buf ^ 1;
        const int kv0 = it * 32;

        // ---- stage tile it+1 into the other buffer (overlaps compute) ----
        if (it + 1 < NT) {
            if (wave == 0)
                tdm_load_k_tile(kb + (kvbase + kv0 + 32) * DM_ + hcol,
                                (unsigned)(uintptr_t)&klds[nbuf][0]);
            U4 vv;
            vv.u4 = *(const uint4*)(vb + (kvbase + kv0 + 32 + ldr) * DM_ + hcol + ldc);
#pragma unroll
            for (int e = 0; e < 8; ++e) vT[nbuf][(ldc + e) * 32 + ldr] = vv.us[e];
            if (it + 2 < NT)   // prefetch V two tiles ahead -> global_prefetch_b8
                __builtin_prefetch(vb + (kvbase + kv0 + 64 + ldr) * DM_ + hcol + ldc, 0, 3);
        }

        // ---- scores: two 16-col sub-tiles, K accumulated over Dk=64 ----
        v8f S[2];
#pragma unroll
        for (int sv = 0; sv < 2; ++sv) {
            v8f s = v8zero();
#pragma unroll
            for (int dsl = 0; dsl < 2; ++dsl) {
                FragAB kf;
                const unsigned short* base = &klds[buf][(sv * 16 + lr) * 64 + dsl * 32 + lg * 8];
                kf.u4[0] = *(const uint4*)(base);
                kf.u4[1] = *(const uint4*)(base + 16);
                s = __builtin_amdgcn_wmma_f32_16x16x32_bf16(
                    false, qf[dsl].v, false, kf.v, (short)0, s, false, false);
            }
            S[sv] = s;
        }

        // ---- online softmax (rows split at lane 16, shfl over 16 lanes) ----
        float alpha[8];
#pragma unroll
        for (int j = 0; j < 8; ++j) {
            float s0 = S[0][j] * 0.125f;   // 1/sqrt(64)
            float s1 = S[1][j] * 0.125f;
            float mx = fmaxf(s0, s1);
#pragma unroll
            for (int off = 8; off >= 1; off >>= 1)
                mx = fmaxf(mx, __shfl_xor(mx, off, 16));
            float nm = fmaxf(mrow[j], mx);
            alpha[j] = __expf(mrow[j] - nm);
            float p0 = __expf(s0 - nm);
            float p1 = __expf(s1 - nm);
            S[0][j] = p0; S[1][j] = p1;
            float rs = p0 + p1;
#pragma unroll
            for (int off = 8; off >= 1; off >>= 1)
                rs += __shfl_xor(rs, off, 16);
            lrow[j] = lrow[j] * alpha[j] + rs;
            mrow[j] = nm;
        }
#pragma unroll
        for (int dt = 0; dt < 4; ++dt)
#pragma unroll
            for (int j = 0; j < 8; ++j) o[dt][j] *= alpha[j];

        // ---- P: C-layout -> bf16 A-layout via per-wave LDS ----
        unsigned short* P = Plds[wave];
#pragma unroll
        for (int sv = 0; sv < 2; ++sv)
#pragma unroll
            for (int j = 0; j < 8; ++j)
                P[(j + lg * 8) * 32 + sv * 16 + lr] = f2bf(S[sv][j]);
        FragAB pf;
        {
            const unsigned short* base = &P[lr * 32 + lg * 8];
            pf.u4[0] = *(const uint4*)(base);
            pf.u4[1] = *(const uint4*)(base + 16);
        }

        // ---- o += P * V ----
#pragma unroll
        for (int dt = 0; dt < 4; ++dt) {
            FragAB vf;
            const unsigned short* base = &vT[buf][(dt * 16 + lr) * 32 + lg * 8];
            vf.u4[0] = *(const uint4*)(base);
            vf.u4[1] = *(const uint4*)(base + 16);
            o[dt] = __builtin_amdgcn_wmma_f32_16x16x32_bf16(
                false, pf.v, false, vf.v, (short)0, o[dt], false, false);
        }

        // ---- drain TDM for tile it+1, then block-wide rendezvous ----
        if (it + 1 < NT && wave == 0) __builtin_amdgcn_s_wait_tensorcnt(0);
        __syncthreads();
    }

    // ---- normalize + store bf16 [B*S, H*Dk] ----
#pragma unroll
    for (int dt = 0; dt < 4; ++dt)
#pragma unroll
        for (int j = 0; j < 8; ++j) {
            int row = j + lg * 8;
            float val = o[dt][j] / lrow[j];
            ob[(qrow + row) * DM_ + hcol + dt * 16 + lr] = f2bf(val);
        }
}

// ---------------------------------------------------------------------------
// launch
// ---------------------------------------------------------------------------
extern "C" void kernel_launch(void* const* d_in, const int* in_sizes, int n_in,
                              void* d_out, int out_size, void* d_ws, size_t ws_size,
                              hipStream_t stream) {
    const float* query = (const float*)d_in[0];
    const float* key   = (const float*)d_in[1];
    const float* value = (const float*)d_in[2];
    const float* Wq = (const float*)d_in[3];  const float* bq = (const float*)d_in[4];
    const float* Wk = (const float*)d_in[5];  const float* bk = (const float*)d_in[6];
    const float* Wv = (const float*)d_in[7];  const float* bv = (const float*)d_in[8];
    const float* Wo = (const float*)d_in[9];  const float* bo = (const float*)d_in[10];

    char* ws = (char*)d_ws;
    const size_t SZ_QKV = (size_t)MTOT * DM_ * 2;       // 16 MB each (bf16)
    unsigned short* qb    = (unsigned short*)(ws);
    unsigned short* kb    = (unsigned short*)(ws + SZ_QKV);
    unsigned short* vb    = (unsigned short*)(ws + 2 * SZ_QKV);
    unsigned short* attnb = (unsigned short*)(ws + 3 * SZ_QKV);
    unsigned short* Wq_b  = (unsigned short*)(ws + 4 * SZ_QKV);
    unsigned short* Wk_b  = (unsigned short*)((char*)Wq_b + (size_t)DM_ * DM_ * 2);
    unsigned short* Wv_b  = (unsigned short*)((char*)Wk_b + (size_t)UD_ * DM_ * 2);
    unsigned short* Wo_b  = (unsigned short*)((char*)Wv_b + (size_t)UD_ * DM_ * 2);
    int* kmask = (int*)((char*)Wo_b + (size_t)DM_ * DM_ * 2);
    int* vmask = kmask + MTOT;

    // weights fp32 -> bf16
    {
        int n4 = DM_ * DM_ / 4;
        cvt_f32_bf16<<<(n4 + 255) / 256, 256, 0, stream>>>(Wq, Wq_b, n4);
        cvt_f32_bf16<<<(n4 + 255) / 256, 256, 0, stream>>>(Wo, Wo_b, n4);
        int n4s = UD_ * DM_ / 4;
        cvt_f32_bf16<<<(n4s + 255) / 256, 256, 0, stream>>>(Wk, Wk_b, n4s);
        cvt_f32_bf16<<<(n4s + 255) / 256, 256, 0, stream>>>(Wv, Wv_b, n4s);
    }
    // zero-row masks
    mask_kernel<<<MTOT, 256, 0, stream>>>(key, value, kmask, vmask);

    // projections (fp32 A, bf16 out)
    dim3 g(MTOT / 128, DM_ / 128);
    gemm_proj<false, false><<<g, 256, 0, stream>>>(query, DM_, Wq_b, bq, nullptr, qb);
    gemm_proj<false, false><<<g, 256, 0, stream>>>(key,   UD_, Wk_b, bk, kmask,  kb);
    gemm_proj<false, false><<<g, 256, 0, stream>>>(value, UD_, Wv_b, bv, vmask,  vb);

    // attention (TDM-staged K, double-buffered)
    attn_kernel<<<dim3(S_ / 128, B_ * H_), 256, 0, stream>>>(qb, kb, vb, attnb);

    // output projection (bf16 A, fp32 out)
    gemm_proj<true, true><<<g, 256, 0, stream>>>(attnb, DM_, Wo_b, bo, nullptr, (float*)d_out);
}
